// MLAAttention_69836168233263
// MI455X (gfx1250) — matrix operand order repaired and checked by reading, not compiled, not run on hardware
//
#include <hip/hip_runtime.h>
#include <hip/hip_bf16.h>
#include <math.h>

#define BQ   2
#define LSEQ 2048
#define DM   1024
#define NH   16
#define HD   64
#define DC   128
#define MT   (BQ*LSEQ)   // 4096 total rows

typedef __attribute__((ext_vector_type(8)))  __bf16 v8bf;
typedef __attribute__((ext_vector_type(16))) __bf16 v16bf;
typedef __attribute__((ext_vector_type(8)))  float  v8f;

// ---- WMMA fragment helpers (gfx1250 wave32; this code's contiguous k order, same in A and B) ----

// A operand 16x32 bf16: lane<16 -> row=lane, elems {K0..15};
//                       lane>=16 -> row=lane-16, elems {K16..31}
__device__ __forceinline__ v16bf ld_a_frag(const __bf16* rowk0, int koff) {
  v8bf lo = *(const v8bf*)(rowk0 + 2 * koff);
  v8bf hi = *(const v8bf*)(rowk0 + 2 * koff + 8);
  return __builtin_shufflevector(lo, hi, 0,1,2,3,4,5,6,7,8,9,10,11,12,13,14,15);
}

// B operand 32x16 bf16: lane n holds column n; lanes 0-15: K0..15, lanes 16-31: K16..31
// (contiguous per lane -> single 32B load from an NxK-stored matrix row)
__device__ __forceinline__ v16bf ld_b_frag(const __bf16* rowk0, int half) {
  return *(const v16bf*)(rowk0 + half * 16);
}

__device__ __forceinline__ v8f wmma_bf16f32(v16bf a, v16bf b, v8f c) {
  return __builtin_amdgcn_wmma_f32_16x16x32_bf16(false, a, false, b, (short)0, c,
                                                 false, false);
}

// In-half (16-lane group) xor butterfly via ds_swizzle group-of-32 pattern:
// offset = {0, xor[4:0], or=0, and=0x1f}. Single LDS-unit op, no index VGPR.
#define SWZ_XOR(x, imm) \
  __int_as_float(__builtin_amdgcn_ds_swizzle(__float_as_int(x), (imm)))

__device__ __forceinline__ float red_max16(float x) {
  x = fmaxf(x, SWZ_XOR(x, 0x041f));   // xor 1
  x = fmaxf(x, SWZ_XOR(x, 0x081f));   // xor 2
  x = fmaxf(x, SWZ_XOR(x, 0x101f));   // xor 4
  x = fmaxf(x, SWZ_XOR(x, 0x201f));   // xor 8
  return x;
}
__device__ __forceinline__ float red_sum16(float x) {
  x += SWZ_XOR(x, 0x041f);
  x += SWZ_XOR(x, 0x081f);
  x += SWZ_XOR(x, 0x101f);
  x += SWZ_XOR(x, 0x201f);
  return x;
}

// ---------------------------------------------------------------------------
// Elementwise fp32 -> bf16 convert
__global__ void cvt_f32_bf16(const float* __restrict__ src, __bf16* __restrict__ dst, int n) {
  int i = blockIdx.x * blockDim.x + threadIdx.x;
  if (i < n) dst[i] = (__bf16)src[i];
}

// Transpose + convert: src KxN fp32 row-major -> dst NxK bf16 row-major
__global__ void tr_f32_bf16(const float* __restrict__ src, __bf16* __restrict__ dst,
                            int K, int N) {
  int i = blockIdx.x * blockDim.x + threadIdx.x;
  if (i < K * N) {
    int k = i / N, n = i % N;
    dst[(size_t)n * K + k] = (__bf16)src[i];
  }
}

// ---------------------------------------------------------------------------
// GEMM: Out[MxN] = A[MxK] * W, with W stored transposed as Bt[NxK], all bf16,
// fp32 accumulate via v_wmma_f32_16x16x32_bf16. 4 waves/block, each wave one
// 16x16 tile of a 16x64 block (A frag re-fetch hits WGP$).
template <bool OUT_F32>
__global__ __launch_bounds__(128) void
gemm_wmma(const __bf16* __restrict__ A, const __bf16* __restrict__ Bt,
          void* __restrict__ Out, int M, int N, int K) {
  const int lane = threadIdx.x & 31;
  const int wave = threadIdx.x >> 5;
  const int half = lane >> 4;
  const int lr   = lane & 15;
  const int koff = half * 8;
  const int m0 = blockIdx.x * 16;
  const int n0 = blockIdx.y * 64 + wave * 16;

  const __bf16* arow = A  + (size_t)(m0 + lr) * K;
  const __bf16* brow = Bt + (size_t)(n0 + lr) * K;

  v8f acc = {};
  for (int k0 = 0; k0 < K; k0 += 32) {
    v16bf a = ld_a_frag(arow + k0, koff);
    v16bf b = ld_b_frag(brow + k0, half);
    acc = wmma_bf16f32(a, b, acc);
  }

  // C/D layout: elem i -> row m0 + i + half*8, col n0 + lr
  const int mrow = m0 + half * 8;
  if (OUT_F32) {
    float* O = (float*)Out;
    #pragma unroll
    for (int i = 0; i < 8; ++i) O[(size_t)(mrow + i) * N + n0 + lr] = acc[i];
  } else {
    __bf16* O = (__bf16*)Out;
    #pragma unroll
    for (int i = 0; i < 8; ++i) O[(size_t)(mrow + i) * N + n0 + lr] = (__bf16)acc[i];
  }
}

// ---------------------------------------------------------------------------
// RoPE-variant + head regrouping:
//   q'  = q*cos_q + rot(q)*cos_k      (reference uses cos for both terms)
//   k'  = k*cos_q + rot(k)*cos_k
//   Qh[b,h,l,d]  = q' * 1/sqrt(64)
//   Ks[b,h,l,d]  = k_c + k'          (folds the two QK einsums into one)
//   Vt[b,h,d,l]  = v_c               (transposed for PV B-operand)
__global__ void rope_prep(const __bf16* __restrict__ qr, const __bf16* __restrict__ kr,
                          const __bf16* __restrict__ kc, const __bf16* __restrict__ vc,
                          const float* __restrict__ tsq, const float* __restrict__ tskv,
                          __bf16* __restrict__ Qh, __bf16* __restrict__ Ks,
                          __bf16* __restrict__ Vt) {
  const int bl = blockIdx.x;            // 0..B*L-1
  const int b  = bl / LSEQ;
  const int l  = bl % LSEQ;
  const int d  = threadIdx.x;           // 0..63
  const int i  = d & 31;
  const float freq = powf(10000.0f, -(float)(2 * i) / (float)HD);
  const float cq = cosf(tsq[bl]  * freq);
  const float ck = cosf(tskv[bl] * freq);
  const int dpart = (d < 32) ? d + 32 : d - 32;
  const float scale = 0.125f;           // 64^-0.5

  for (int h = 0; h < NH; ++h) {
    const size_t base = (size_t)bl * DM + h * HD;
    float q  = (float)qr[base + d];
    float qp = (float)qr[base + dpart];
    float qrot = (d < 32) ? -qp : qp;
    float qv = q * cq + qrot * ck;

    float k  = (float)kr[base + d];
    float kp = (float)kr[base + dpart];
    float krot = (d < 32) ? -kp : kp;
    float kv = k * cq + krot * ck;

    float kcv = (float)kc[base + d];
    float vcv = (float)vc[base + d];

    const size_t bh = (size_t)b * NH + h;
    const size_t hbase = (bh * LSEQ + l) * HD + d;
    Qh[hbase] = (__bf16)(qv * scale);
    Ks[hbase] = (__bf16)(kcv + kv);
    Vt[(bh * HD + d) * LSEQ + l] = (__bf16)vcv;
  }
}

// ---------------------------------------------------------------------------
// Flash attention: one wave per (b,h,16-query tile). Online softmax over 2048
// keys in chunks of 32. S=Q*K^T via WMMA, P->bf16 through LDS (in-order DS
// pipe + s_wait_dscnt) to re-fragment C-layout into A-layout, O += P*V.
// launch_bounds(32,1): one fat wave, full VGPR budget -> no spills.
__global__ __launch_bounds__(32, 1) void
flash_attn(const __bf16* __restrict__ Qh, const __bf16* __restrict__ Ks,
           const __bf16* __restrict__ Vt, __bf16* __restrict__ Out) {
  __shared__ __attribute__((aligned(16))) __bf16 pLDS[16][40]; // padded rows

  const int lane = threadIdx.x & 31;
  const int half = lane >> 4;
  const int lr   = lane & 15;
  const int koff = half * 8;
  const int bh = blockIdx.y;            // 0..B*H-1
  const int b  = bh / NH;
  const int h  = bh % NH;
  const int q0 = blockIdx.x * 16;

  const __bf16* Qb = Qh + ((size_t)bh * LSEQ + q0) * HD;   // 16 x 64
  const __bf16* Kb = Ks + (size_t)bh * LSEQ * HD;          // L x 64 (rows = keys)
  const __bf16* Vb = Vt + (size_t)bh * HD * LSEQ;          // 64 x L (rows = dims)

  // Q A-operand frags (d-chunks 0..31 and 32..63), loaded once
  v16bf aq0 = ld_a_frag(Qb + (size_t)lr * HD + 0,  koff);
  v16bf aq1 = ld_a_frag(Qb + (size_t)lr * HD + 32, koff);

  // hoisted, pointer-bumped row pointers (no per-iter 64-bit address math)
  const __bf16* pk0 = Kb + (size_t)(lr)      * HD;   // key tile 0 row
  const __bf16* pk1 = Kb + (size_t)(16 + lr) * HD;   // key tile 1 row
  const __bf16* pv0 = Vb + (size_t)( 0 + lr) * LSEQ; // V dim rows (invariant)
  const __bf16* pv1 = Vb + (size_t)(16 + lr) * LSEQ;
  const __bf16* pv2 = Vb + (size_t)(32 + lr) * LSEQ;
  const __bf16* pv3 = Vb + (size_t)(48 + lr) * LSEQ;

  float m[8], lsum[8];
  v8f o0 = {}, o1 = {}, o2 = {}, o3 = {};
  #pragma unroll
  for (int i = 0; i < 8; ++i) { m[i] = -1e30f; lsum[i] = 0.0f; }

  for (int k0 = 0; k0 < LSEQ;
       k0 += 32, pk0 += 32 * HD, pk1 += 32 * HD,
       pv0 += 32, pv1 += 32, pv2 += 32, pv3 += 32) {
    __builtin_prefetch(pk0 + 32 * HD, 0, 1);   // next chunk -> global_prefetch_b8

    // S tiles: 16 queries x 32 keys, contraction over d=64 (2 wmma per tile)
    v8f s0 = {}, s1 = {};
    s0 = wmma_bf16f32(aq0, ld_b_frag(pk0 + 0,  half), s0);
    s0 = wmma_bf16f32(aq1, ld_b_frag(pk0 + 32, half), s0);
    s1 = wmma_bf16f32(aq0, ld_b_frag(pk1 + 0,  half), s1);
    s1 = wmma_bf16f32(aq1, ld_b_frag(pk1 + 32, half), s1);

    // per-row running max update (rows split by half-wave)
    float alpha[8];
    #pragma unroll
    for (int i = 0; i < 8; ++i) {
      float mc = red_max16(fmaxf(s0[i], s1[i]));
      float mn = fmaxf(m[i], mc);
      alpha[i] = __expf(m[i] - mn);
      m[i] = mn;
    }

    // P = exp(S - m): stash to LDS row-major for A-operand re-fragmenting,
    // and accumulate the row sums
    #pragma unroll
    for (int i = 0; i < 8; ++i) {
      float p0 = __expf(s0[i] - m[i]);
      float p1 = __expf(s1[i] - m[i]);
      pLDS[i + half * 8][lr]      = (__bf16)p0;
      pLDS[i + half * 8][16 + lr] = (__bf16)p1;
      lsum[i] = lsum[i] * alpha[i] + red_sum16(p0 + p1);
    }

    // rescale running O
    #pragma unroll
    for (int i = 0; i < 8; ++i) {
      o0[i] *= alpha[i]; o1[i] *= alpha[i]; o2[i] *= alpha[i]; o3[i] *= alpha[i];
    }

    // DS pipe is in-order within a wave; wait then re-read P as A operand
    asm volatile("s_wait_dscnt 0" ::: "memory");
    v8bf plo = *(const v8bf*)&pLDS[lr][2 * koff];
    v8bf phi = *(const v8bf*)&pLDS[lr][2 * koff + 8];
    v16bf pa = __builtin_shufflevector(plo, phi, 0,1,2,3,4,5,6,7,8,9,10,11,12,13,14,15);

    // O(16x64) += P(16x32) * V(32x64); V B-operand rows from transposed Vt
    o0 = wmma_bf16f32(pa, ld_b_frag(pv0, half), o0);
    o1 = wmma_bf16f32(pa, ld_b_frag(pv1, half), o1);
    o2 = wmma_bf16f32(pa, ld_b_frag(pv2, half), o2);
    o3 = wmma_bf16f32(pa, ld_b_frag(pv3, half), o3);
    asm volatile("s_wait_dscnt 0" ::: "memory"); // LDS reads done before next-iter stores
  }

  // normalize and write attention output regrouped to (B*L) x DM row-major
  #pragma unroll
  for (int i = 0; i < 8; ++i) {
    float inv = 1.0f / lsum[i];
    size_t row = (size_t)b * LSEQ + q0 + i + half * 8;
    size_t base = row * DM + (size_t)h * HD + lr;
    Out[base +  0] = (__bf16)(o0[i] * inv);
    Out[base + 16] = (__bf16)(o1[i] * inv);
    Out[base + 32] = (__bf16)(o2[i] * inv);
    Out[base + 48] = (__bf16)(o3[i] * inv);
  }
}

// ---------------------------------------------------------------------------
extern "C" void kernel_launch(void* const* d_in, const int* in_sizes, int n_in,
                              void* d_out, int out_size, void* d_ws, size_t ws_size,
                              hipStream_t stream) {
  (void)in_sizes; (void)n_in; (void)out_size; (void)ws_size;
  const float* query  = (const float*)d_in[0];
  const float* tsq    = (const float*)d_in[1];
  const float* tskv   = (const float*)d_in[2];
  const float* w_kv_c = (const float*)d_in[3];
  const float* w_kc   = (const float*)d_in[4];
  const float* w_vc   = (const float*)d_in[5];
  const float* w_qr   = (const float*)d_in[6];
  const float* w_kr   = (const float*)d_in[7];
  const float* w_o    = (const float*)d_in[8];

  __bf16* p = (__bf16*)d_ws;
  __bf16* qb    = p; p += (size_t)MT * DM;
  __bf16* wkvcT = p; p += (size_t)DC * DM;   // [N=128][K=1024]
  __bf16* wkcT  = p; p += (size_t)DM * DC;   // [N=1024][K=128]
  __bf16* wvcT  = p; p += (size_t)DM * DC;
  __bf16* wqrT  = p; p += (size_t)DM * DM;
  __bf16* wkrT  = p; p += (size_t)DM * DM;
  __bf16* woT   = p; p += (size_t)DM * DM;
  __bf16* ckv   = p; p += (size_t)MT * DC;
  __bf16* kc    = p; p += (size_t)MT * DM;
  __bf16* vc    = p; p += (size_t)MT * DM;
  __bf16* qr    = p; p += (size_t)MT * DM;
  __bf16* kr    = p; p += (size_t)MT * DM;
  __bf16* Qh    = p; p += (size_t)MT * DM;
  __bf16* Ksum  = p; p += (size_t)MT * DM;
  __bf16* Vt    = p; p += (size_t)MT * DM;
  __bf16* attno = p; p += (size_t)MT * DM;

  const int thr = 256;
  cvt_f32_bf16<<<(MT * DM + thr - 1) / thr, thr, 0, stream>>>(query, qb, MT * DM);
  tr_f32_bf16<<<(DM * DC + thr - 1) / thr, thr, 0, stream>>>(w_kv_c, wkvcT, DM, DC);
  tr_f32_bf16<<<(DC * DM + thr - 1) / thr, thr, 0, stream>>>(w_kc,   wkcT,  DC, DM);
  tr_f32_bf16<<<(DC * DM + thr - 1) / thr, thr, 0, stream>>>(w_vc,   wvcT,  DC, DM);
  tr_f32_bf16<<<(DM * DM + thr - 1) / thr, thr, 0, stream>>>(w_qr,   wqrT,  DM, DM);
  tr_f32_bf16<<<(DM * DM + thr - 1) / thr, thr, 0, stream>>>(w_kr,   wkrT,  DM, DM);
  tr_f32_bf16<<<(DM * DM + thr - 1) / thr, thr, 0, stream>>>(w_o,    woT,   DM, DM);

  // projections (all WMMA)
  gemm_wmma<false><<<dim3(MT / 16, DC / 64), 128, 0, stream>>>(qb,  wkvcT, ckv, MT, DC, DM);
  gemm_wmma<false><<<dim3(MT / 16, DM / 64), 128, 0, stream>>>(ckv, wkcT,  kc,  MT, DM, DC);
  gemm_wmma<false><<<dim3(MT / 16, DM / 64), 128, 0, stream>>>(ckv, wvcT,  vc,  MT, DM, DC);
  gemm_wmma<false><<<dim3(MT / 16, DM / 64), 128, 0, stream>>>(qb,  wqrT,  qr,  MT, DM, DM);
  gemm_wmma<false><<<dim3(MT / 16, DM / 64), 128, 0, stream>>>(qb,  wkrT,  kr,  MT, DM, DM);

  rope_prep<<<MT, HD, 0, stream>>>(qr, kr, kc, vc, tsq, tskv, Qh, Ksum, Vt);

  flash_attn<<<dim3(LSEQ / 16, BQ * NH), 32, 0, stream>>>(Qh, Ksum, Vt, attno);

  // output projection, fp32 result straight into d_out
  gemm_wmma<true><<<dim3(MT / 16, DM / 64), 128, 0, stream>>>(attno, woT, d_out, MT, DM, DM);
}
